// InvSlotAttentionGuide_49770081026760
// MI455X (gfx1250) — matrix operand order repaired and checked by reading, not compiled, not run on hardware
//
#include <hip/hip_runtime.h>

typedef __attribute__((ext_vector_type(16))) _Float16 v16h;
typedef __attribute__((ext_vector_type(4)))  _Float16 h4v;
typedef __attribute__((ext_vector_type(2)))  _Float16 h2v;
typedef __attribute__((ext_vector_type(8)))  float    v8f;

#define NB    16
#define DIM   64
#define NSLOT 7
#define NPIX  1024
#define EPSF  1e-8f
#define USE_ASYNC 1

// ---------------- wave32 helpers ----------------
__device__ inline float wave_sum(float v) {
#pragma unroll
  for (int m = 16; m >= 1; m >>= 1) v += __shfl_xor(v, m, 32);
  return v;
}
__device__ inline float wave_max(float v) {
#pragma unroll
  for (int m = 16; m >= 1; m >>= 1) v = fmaxf(v, __shfl_xor(v, m, 32));
  return v;
}

// ---------------- CDNA5 async memory->LDS copy (16B per lane) ----------------
__device__ inline void async_b128(void* lds_dst, const _Float16* gbase, unsigned byte_off) {
#if USE_ASYNC
  unsigned lds_off = (unsigned)(unsigned long long)lds_dst;
  asm volatile("global_load_async_to_lds_b128 %0, %1, %2 offset:0"
               :: "v"(lds_off), "v"(byte_off), "s"((unsigned long long)gbase)
               : "memory");
#else
  *(uint4*)lds_dst = *(const uint4*)((const char*)gbase + byte_off);
#endif
}
__device__ inline void async_wait() {
#if USE_ASYNC
  asm volatile("s_wait_asynccnt 0x0" ::: "memory");
#endif
}

// ---------------- WMMA fragment loaders (CDNA5 ISA 7.12.2 layouts) ----------------
__device__ inline v16h load_a_frag(const _Float16* base, int stride, int lane) {
  int m = lane & 15;
  int hi8 = (lane >> 4) << 3;
  const _Float16* rowp = base + m * stride;
  v16h a;
#pragma unroll
  for (int v = 0; v < 8; ++v) {
    int k = ((v >= 4) ? 16 : 0) + hi8 + ((v & 3) << 1);
    h2v p = *(const h2v*)(rowp + k);
    a[2 * v] = p[0];
    a[2 * v + 1] = p[1];
  }
  return a;
}
__device__ inline v16h load_b_frag(const _Float16* base, int stride, int lane) {
  int n = lane & 15;
  int khi = (lane >> 4) << 3;
  const _Float16* rowp = base + n * stride + khi;
  v16h b;
#pragma unroll
  for (int v = 0; v < 8; ++v) {
    h2v p = *(const h2v*)(rowp + ((v >= 4) ? 16 : 0) + 2 * (v & 3));
    b[2 * v] = p[0];
    b[2 * v + 1] = p[1];
  }
  return b;
}
__device__ inline v8f wmma_f16(v16h a, v16h b, v8f c) {
  return __builtin_amdgcn_wmma_f32_16x16x32_f16(false, a, false, b, (short)0, c, false, false);
}

// ---------------- prep kernels (once per launch; tiny) ----------------
__global__ void cvt_f32_f16_kernel(const float* __restrict__ src, _Float16* __restrict__ dst, int n) {
  int i = blockIdx.x * 256 + threadIdx.x;
  if (i < n) dst[i] = (_Float16)src[i];
}
// conv weights -> f16, K padded to multiple of 32 with zeros
__global__ void prep_convw_kernel(const float* __restrict__ src, _Float16* __restrict__ dst,
                                  int Ktot, int Kpad) {
  int i = blockIdx.x * 256 + threadIdx.x;
  if (i >= 64 * Kpad) return;
  int r = i / Kpad, k = i - r * Kpad;
  dst[i] = (k < Ktot) ? (_Float16)src[r * Ktot + k] : (_Float16)0.0f;
}
// GEMM B -> f16 Bt[N][K]; transB: src already [N][K], else src is [K][N]
__global__ void prep_bt_kernel(const float* __restrict__ src, _Float16* __restrict__ dst,
                               int N, int K, int transB) {
  int i = blockIdx.x * 256 + threadIdx.x;
  if (i >= N * K) return;
  int n = i / K, k = i - n * K;
  dst[i] = (_Float16)(transB ? src[n * K + k] : src[k * N + n]);
}

// ---------------- implicit-GEMM conv 5x5 pad2 + bias + relu (f16 in/out) ----------------
template <int CIN, int STRIDE>
__global__ void __launch_bounds__(256) conv_relu_wmma(
    const _Float16* __restrict__ xin, const _Float16* __restrict__ w16,
    const float* __restrict__ bias, _Float16* __restrict__ out,
    int Bn, int H, int W, int Ho, int Wo) {
  constexpr int Ktot = CIN * 25;
  constexpr int Kpad = (Ktot + 31) & ~31;
  constexpr int NKT = Kpad / 32;
  constexpr bool KMASK = (Kpad != Ktot);
  __shared__ _Float16 As[64 * 40];
  __shared__ _Float16 Bs[64 * 40];
  const int HoWo = Ho * Wo;
  const int Npixels = Bn * HoWo;
  const int t = threadIdx.x;
  const int wave = t >> 5, lane = t & 31;
  const int wgpix = blockIdx.x * 64;

  const int srow = t >> 2;       // 0..63: co for A, local pixel for B
  const int sk0 = (t & 3) * 8;   // k offset within 32-tile

  int pix = wgpix + srow;
  bool pvalid = pix < Npixels;
  int pixc = pvalid ? pix : 0;
  int pb = pixc / HoWo;
  int rem = pixc - pb * HoWo;
  int poy = rem / Wo;
  int pox = rem - poy * Wo;

  const int ct = wave & 3;
  const int pt0 = (wave >> 2) * 2;
  v8f c0 = {0, 0, 0, 0, 0, 0, 0, 0};
  v8f c1 = {0, 0, 0, 0, 0, 0, 0, 0};

  for (int kt = 0; kt < NKT; ++kt) {
    const int kbase = kt * 32;
    // stage A (weights, padded f16): contiguous 16B per thread, async to LDS
    async_b128(&As[srow * 40 + sk0], w16, (unsigned)((srow * Kpad + kbase + sk0) * 2));
    // stage B (im2col patch): branchless clamped loads
#pragma unroll
    for (int j = 0; j < 8; ++j) {
      int gk = kbase + sk0 + j;
      int ci = gk / 25;
      int rr = gk - ci * 25;
      int ky = rr / 5;
      int kx = rr - ky * 5;
      int iy = poy * STRIDE + ky - 2;
      int ix = pox * STRIDE + kx - 2;
      bool ok = pvalid & (iy >= 0) & (iy < H) & (ix >= 0) & (ix < W);
      if (KMASK) ok = ok & (gk < Ktot);
      int idx = ok ? ((pb * CIN + ci) * H + iy) * W + ix : 0;
      _Float16 hv = xin[idx];
      Bs[srow * 40 + sk0 + j] = ok ? hv : (_Float16)0.0f;
    }
    async_wait();
    __syncthreads();
    v16h a = load_a_frag(&As[ct * 16 * 40], 40, lane);
    v16h b0 = load_b_frag(&Bs[(pt0 * 16) * 40], 40, lane);
    v16h b1 = load_b_frag(&Bs[((pt0 + 1) * 16) * 40], 40, lane);
    c0 = wmma_f16(a, b0, c0);
    c1 = wmma_f16(a, b1, c1);
    __syncthreads();
  }

  int hi = lane >> 4, col = lane & 15;
#pragma unroll
  for (int tile = 0; tile < 2; ++tile) {
    v8f cc = tile ? c1 : c0;
    int gp = wgpix + (pt0 + tile) * 16 + col;
    if (gp < Npixels) {
      int bb = gp / HoWo;
      int rem2 = gp - bb * HoWo;
      int oy = rem2 / Wo;
      int ox = rem2 - oy * Wo;
#pragma unroll
      for (int r = 0; r < 8; ++r) {
        int co = ct * 16 + r + hi * 8;
        float v = cc[r] + bias[co];
        out[((bb * 64 + co) * Ho + oy) * Wo + ox] = (_Float16)fmaxf(v, 0.0f);
      }
    }
  }
}

// ---------------- generic GEMM: out = act(A_h[M,K] @ Bt_h + bias) (+res) ----------------
// OUTMODE: 0=f32, 1=f16, 2=both, 3=f16 batch-transposed (vT[b][n][row%1024])
template <int K, int N, int OUTMODE, int ACT>
__global__ void __launch_bounds__(256) gemm_wmma(
    const _Float16* __restrict__ A, const _Float16* __restrict__ Bt,
    const float* __restrict__ bias, const float* __restrict__ res,
    float* __restrict__ out32, _Float16* __restrict__ out16, int M) {
  constexpr int LDK = K + 8;
  __shared__ _Float16 Asl[128 * LDK];
  __shared__ _Float16 Btl[N * LDK];
  const int t = threadIdx.x;
  const int wave = t >> 5, lane = t & 31;
  const int rb = blockIdx.x * 128;
  constexpr int KCH = K / 8;  // 16B chunks per row

  // stage Bt (contiguous f16) via async
#pragma unroll
  for (int c = t; c < N * KCH; c += 256) {
    int n = c / KCH;
    int kc = (c - n * KCH) * 8;
    async_b128(&Btl[n * LDK + kc], Bt, (unsigned)((n * K + kc) * 2));
  }
  // stage A rows (clamped; padding rows produce discarded output rows) via async
#pragma unroll
  for (int c = t; c < 128 * KCH; c += 256) {
    int r = c / KCH;
    int kc = (c - r * KCH) * 8;
    int gr = rb + r;
    int grc = (gr < M) ? gr : (M - 1);
    async_b128(&Asl[r * LDK + kc], A, (unsigned)((grc * K + kc) * 2));
  }
  async_wait();
  __syncthreads();

  const int row0 = wave * 16;
  const int hi = lane >> 4, col = lane & 15;
#pragma unroll
  for (int nt = 0; nt < (N >> 4); ++nt) {
    float bv = bias ? bias[nt * 16 + col] : 0.0f;
    v8f c = {bv, bv, bv, bv, bv, bv, bv, bv};
#pragma unroll
    for (int kt = 0; kt < (K >> 5); ++kt) {
      v16h a = load_a_frag(&Asl[row0 * LDK + kt * 32], LDK, lane);
      v16h b = load_b_frag(&Btl[(nt * 16) * LDK + kt * 32], LDK, lane);
      c = wmma_f16(a, b, c);
    }
#pragma unroll
    for (int r = 0; r < 8; ++r) {
      int gr = rb + row0 + r + hi * 8;
      if (gr < M) {
        int gc = nt * 16 + col;
        float v = c[r];
        if (res) v += res[gr * N + gc];
        if (ACT == 1) v = fmaxf(v, 0.0f);
        if (OUTMODE == 0 || OUTMODE == 2) out32[gr * N + gc] = v;
        if (OUTMODE == 1 || OUTMODE == 2) out16[gr * N + gc] = (_Float16)v;
        if (OUTMODE == 3) {
          int bb = gr >> 10, rr = gr & 1023;
          out16[(bb * 64 + gc) * 1024 + rr] = (_Float16)v;
        }
      }
    }
  }
}

// ---------------- positional embedding (f16 -> f16) ----------------
__global__ void pos_embed_kernel(const _Float16* __restrict__ h4, const float* __restrict__ pw,
                                 const float* __restrict__ pb, _Float16* __restrict__ feats0) {
  int idx = blockIdx.x * blockDim.x + threadIdx.x;
  if (idx >= NB * NPIX * DIM) return;
  int c = idx & 63;
  int n = (idx >> 6) & 1023;
  int b = idx >> 16;
  int y = n >> 5, x = n & 31;
  float fy = (float)y * (1.0f / 31.0f), fx = (float)x * (1.0f / 31.0f);
  float v = (float)h4[((b * DIM + c) * 32 + y) * 32 + x];
  v += fy * pw[c] + fx * pw[DIM + c] + (1.0f - fy) * pw[2 * DIM + c] +
       (1.0f - fx) * pw[3 * DIM + c] + pb[c];
  feats0[idx] = (_Float16)v;
}

// ---------------- layernorm (wave/row) -> f16 out, optional marginal logit ----------------
__global__ void __launch_bounds__(256) ln_rows_kernel(
    const float* __restrict__ xin, const float* __restrict__ g, const float* __restrict__ bta,
    const float* __restrict__ mw, const float* __restrict__ mb,
    _Float16* __restrict__ out16, float* __restrict__ logits, int Nrows) {
  int wave = threadIdx.x >> 5, lane = threadIdx.x & 31;
  int row = blockIdx.x * 8 + wave;
  if (row >= Nrows) return;
  const float* xr = xin + row * DIM;
  float x0 = xr[lane], x1 = xr[lane + 32];
  float m = wave_sum(x0 + x1) * (1.0f / 64.0f);
  float d0 = x0 - m, d1 = x1 - m;
  float var = wave_sum(d0 * d0 + d1 * d1) * (1.0f / 64.0f);
  float inv = rsqrtf(var + 1e-5f);
  float y0 = d0 * inv * g[lane] + bta[lane];
  float y1 = d1 * inv * g[lane + 32] + bta[lane + 32];
  out16[row * DIM + lane] = (_Float16)y0;
  out16[row * DIM + lane + 32] = (_Float16)y1;
  if (mw) {
    float dot = wave_sum(y0 * mw[lane] + y1 * mw[lane + 32]);
    if (lane == 0) logits[row] = dot + mb[0];
  }
}

// ---------------- per-batch softmax * scale ----------------
__global__ void __launch_bounds__(256) softmax_scale_kernel(
    const float* __restrict__ logits, float* __restrict__ outp, int L, float scale) {
  __shared__ float red[8];
  int b = blockIdx.x;
  int t = threadIdx.x, wave = t >> 5, lane = t & 31;
  const float* lp = logits + b * L;
  float mx = -1e30f;
  for (int i = t; i < L; i += 256) mx = fmaxf(mx, lp[i]);
  mx = wave_max(mx);
  if (lane == 0) red[wave] = mx;
  __syncthreads();
  float bm = red[0];
#pragma unroll
  for (int w = 1; w < 8; ++w) bm = fmaxf(bm, red[w]);
  __syncthreads();
  float sum = 0.0f;
  for (int i = t; i < L; i += 256) sum += __expf(lp[i] - bm);
  sum = wave_sum(sum);
  if (lane == 0) red[wave] = sum;
  __syncthreads();
  float tot = 0.0f;
#pragma unroll
  for (int w = 0; w < 8; ++w) tot += red[w];
  float invt = scale / tot;
  for (int i = t; i < L; i += 256) outp[b * L + i] = __expf(lp[i] - bm) * invt;
}

__global__ void slots_init_kernel(const float* __restrict__ mu, const float* __restrict__ ls,
                                  const float* __restrict__ noise, float* __restrict__ slots,
                                  _Float16* __restrict__ slots_h) {
  int idx = blockIdx.x * blockDim.x + threadIdx.x;
  if (idx >= NB * NSLOT * DIM) return;
  int c = idx & 63;
  float v = mu[c] + __expf(ls[c]) * noise[idx];
  slots[idx] = v;
  slots_h[idx] = (_Float16)v;
}

__global__ void gru_kernel(const float* __restrict__ gi, const float* __restrict__ gh,
                           const float* __restrict__ h, float* __restrict__ outp) {
  int idx = blockIdx.x * blockDim.x + threadIdx.x;
  if (idx >= NB * NSLOT * DIM) return;
  int row = idx >> 6, c = idx & 63;
  const float* gir = gi + row * 192;
  const float* ghr = gh + row * 192;
  float r = 1.0f / (1.0f + __expf(-(gir[c] + ghr[c])));
  float z = 1.0f / (1.0f + __expf(-(gir[c + 64] + ghr[c + 64])));
  float n = tanhf(gir[c + 128] + r * ghr[c + 128]);
  outp[idx] = (1.0f - z) * n + z * h[idx];
}

// ---------------- updates = attn_t[7,1024] @ v[1024,64] per batch (v pre-transposed) ----------------
__global__ void __launch_bounds__(256) attn_v_wmma(
    const float* __restrict__ attn_t, const _Float16* __restrict__ vT,
    _Float16* __restrict__ updates16) {
  __shared__ _Float16 As[16 * 40];
  __shared__ _Float16 Bs[64 * 40];
  int b = blockIdx.x;
  int t = threadIdx.x, wave = t >> 5, lane = t & 31;
  const float* ab = attn_t + b * NSLOT * NPIX;
  const _Float16* vb = vT + b * DIM * NPIX;
  v8f c = {0, 0, 0, 0, 0, 0, 0, 0};
  for (int kt = 0; kt < NPIX / 32; ++kt) {
    const int kb = kt * 32;
    // stage B: vT rows contiguous, async 16B per thread
    {
      int n = t >> 2, kc = (t & 3) * 8;
      async_b128(&Bs[n * 40 + kc], vb, (unsigned)((n * NPIX + kb + kc) * 2));
    }
    // stage A: attn rows (f32 -> f16), rows >= 7 zeroed branchlessly
    if (t < 128) {
      int s = t >> 3, kc = (t & 7) * 4;
      int sc = (s < NSLOT) ? s : (NSLOT - 1);
      float4 av = *(const float4*)(ab + sc * NPIX + kb + kc);
      float msk = (s < NSLOT) ? 1.0f : 0.0f;
      h4v hv;
      hv[0] = (_Float16)(av.x * msk);
      hv[1] = (_Float16)(av.y * msk);
      hv[2] = (_Float16)(av.z * msk);
      hv[3] = (_Float16)(av.w * msk);
      *(h4v*)&As[s * 40 + kc] = hv;
    }
    async_wait();
    __syncthreads();
    if (wave < 4) {
      v16h a = load_a_frag(As, 40, lane);
      v16h bf = load_b_frag(&Bs[(wave * 16) * 40], 40, lane);
      c = wmma_f16(a, bf, c);
    }
    __syncthreads();
  }
  if (wave < 4) {
    int hi = lane >> 4, col = lane & 15;
#pragma unroll
    for (int r = 0; r < 8; ++r) {
      int s = r + hi * 8;
      if (s < NSLOT) updates16[(b * NSLOT + s) * DIM + wave * 16 + col] = (_Float16)c[r];
    }
  }
}

// ---------------- cdist + MESH (grad through 5 Sinkhorn iters) + final Sinkhorn, all in LDS ----------------
#define CS(n, s) Cs[(n) * 7 + (s)]
#define DM(n, s) dMs[(n) * 7 + (s)]
__global__ void __launch_bounds__(256) mesh_sinkhorn_kernel(
    const float* __restrict__ kfeat, const float* __restrict__ qbuf,
    const float* __restrict__ abuf, const float* __restrict__ bmbuf,
    float* __restrict__ attn_t) {
  __shared__ float Cs[NPIX * 7];
  __shared__ float dMs[NPIX * 7];
  __shared__ float fl[NPIX];
  __shared__ float fst[5][NPIX];
  __shared__ float la[NPIX];
  __shared__ float dfl[NPIX];
  __shared__ float qs[7][DIM];
  __shared__ float qsq[8];
  __shared__ float lb[8];
  __shared__ float gl[8];
  __shared__ float gst[6][8];
  __shared__ float dgl[8];
  __shared__ float red[8][8];

  int b = blockIdx.x;
  int t = threadIdx.x, wave = t >> 5, lane = t & 31;

  for (int e = t; e < 7 * DIM; e += 256) qs[e >> 6][e & 63] = qbuf[b * 7 * DIM + e];
  if (t < 7) {
    lb[t] = logf(bmbuf[b * 7 + t] + EPSF);
    gl[t] = 0.0f;
  }
  __syncthreads();
  if (t < 7) {
    float s2 = 0.f;
    for (int c = 0; c < DIM; ++c) { float qv = qs[t][c]; s2 += qv * qv; }
    qsq[t] = s2;
  }
  __syncthreads();

  for (int i = 0; i < 4; ++i) {
    int n = t + i * 256;
    const float* kr = kfeat + (b * NPIX + n) * DIM;
    float ks = 0.f;
    float dots[7] = {0, 0, 0, 0, 0, 0, 0};
    for (int c = 0; c < DIM; ++c) {
      float kv = kr[c];
      ks += kv * kv;
#pragma unroll
      for (int s = 0; s < 7; ++s) dots[s] += kv * qs[s][c];
    }
#pragma unroll
    for (int s = 0; s < 7; ++s) {
      float d2 = ks + qsq[s] - 2.0f * dots[s];
      CS(n, s) = sqrtf(fmaxf(d2, 1e-12f));
    }
    la[n] = logf(abuf[b * NPIX + n] + EPSF);
  }
  __syncthreads();

  for (int mesh = 0; mesh < 5; ++mesh) {
    bool fin = (mesh == 4);
    if (!fin && t < 7) gst[0][t] = gl[t];
    __syncthreads();
    for (int it = 0; it < 5; ++it) {
      for (int i = 0; i < 4; ++i) {
        int n = t + i * 256;
        float mx = -1e30f;
#pragma unroll
        for (int s = 0; s < 7; ++s) mx = fmaxf(mx, gl[s] - CS(n, s));
        float sum = 0.f;
#pragma unroll
        for (int s = 0; s < 7; ++s) sum += __expf(gl[s] - CS(n, s) - mx);
        float f = la[n] - (mx + logf(sum));
        fl[n] = f;
        if (!fin) fst[it][n] = f;
      }
      __syncthreads();
      float acc[7] = {0, 0, 0, 0, 0, 0, 0};
      for (int i = 0; i < 4; ++i) {
        int n = t + i * 256;
        float f = fl[n];
#pragma unroll
        for (int s = 0; s < 7; ++s) acc[s] += __expf(f - CS(n, s));
      }
#pragma unroll
      for (int s = 0; s < 7; ++s) acc[s] = wave_sum(acc[s]);
      if (lane == 0) {
#pragma unroll
        for (int s = 0; s < 7; ++s) red[wave][s] = acc[s];
      }
      __syncthreads();
      if (t < 7) {
        float tot = 0.f;
#pragma unroll
        for (int w = 0; w < 8; ++w) tot += red[w][t];
        float gnew = lb[t] - logf(tot);
        gl[t] = gnew;
        if (!fin) gst[it + 1][t] = gnew;
      }
      __syncthreads();
    }
    if (fin) break;
    {
      float acc[7] = {0, 0, 0, 0, 0, 0, 0};
      for (int i = 0; i < 4; ++i) {
        int n = t + i * 256;
        float f = fl[n];
        float rs = 0.f;
#pragma unroll
        for (int s = 0; s < 7; ++s) {
          float P = __expf(f + gl[s] - CS(n, s));
          float dP = -(logf(P + EPSF) + P / (P + EPSF)) * (1.0f / 16.0f);
          float dm = dP * P;
          DM(n, s) = dm;
          rs += dm;
          acc[s] += dm;
        }
        dfl[n] = rs;
      }
#pragma unroll
      for (int s = 0; s < 7; ++s) acc[s] = wave_sum(acc[s]);
      if (lane == 0) {
#pragma unroll
        for (int s = 0; s < 7; ++s) red[wave][s] = acc[s];
      }
      __syncthreads();
      if (t < 7) {
        float tot = 0.f;
#pragma unroll
        for (int w = 0; w < 8; ++w) tot += red[w][t];
        dgl[t] = tot;
      }
      __syncthreads();
    }
    for (int it = 4; it >= 0; --it) {
      float accB[7] = {0, 0, 0, 0, 0, 0, 0};
      for (int i = 0; i < 4; ++i) {
        int n = t + i * 256;
        float ft = fst[it][n];
        float df = (it == 4) ? dfl[n] : 0.0f;
#pragma unroll
        for (int s = 0; s < 7; ++s) {
          float Av = __expf(ft + gst[it + 1][s] - CS(n, s) - lb[s]);
          DM(n, s) -= dgl[s] * Av;
          df -= dgl[s] * Av;
        }
#pragma unroll
        for (int s = 0; s < 7; ++s) {
          float Bv = __expf(ft + gst[it][s] - CS(n, s) - la[n]);
          DM(n, s) -= df * Bv;
          accB[s] -= df * Bv;
        }
      }
      __syncthreads();
#pragma unroll
      for (int s = 0; s < 7; ++s) accB[s] = wave_sum(accB[s]);
      if (lane == 0) {
#pragma unroll
        for (int s = 0; s < 7; ++s) red[wave][s] = accB[s];
      }
      __syncthreads();
      if (t < 7) {
        float tot = 0.f;
#pragma unroll
        for (int w = 0; w < 8; ++w) tot += red[w][t];
        dgl[t] = tot;
      }
      __syncthreads();
    }
    for (int i = 0; i < 4; ++i) {
      int n = t + i * 256;
#pragma unroll
      for (int s = 0; s < 7; ++s) CS(n, s) += DM(n, s);
    }
    __syncthreads();
  }

  for (int i = 0; i < 4; ++i) {
    int n = t + i * 256;
    float f = fl[n];
#pragma unroll
    for (int s = 0; s < 7; ++s)
      attn_t[(b * 7 + s) * NPIX + n] = __expf(f + gl[s] - CS(n, s));
  }
}

__global__ void pack_kernel(const float* __restrict__ slots, const float* __restrict__ attn,
                            float* __restrict__ outp) {
  int idx = blockIdx.x * blockDim.x + threadIdx.x;
  const int ns = NB * NSLOT * DIM;
  const int na = NB * NSLOT * NPIX;
  if (idx >= ns + na) return;
  outp[idx] = (idx < ns) ? slots[idx] : attn[idx - ns];
}

// ---------------- host orchestration ----------------
extern "C" void kernel_launch(void* const* d_in, const int* in_sizes, int n_in,
                              void* d_out, int out_size, void* d_ws, size_t ws_size,
                              hipStream_t stream) {
  (void)in_sizes; (void)n_in; (void)out_size; (void)ws_size;
  const float* x        = (const float*)d_in[0];
  const float* noise    = (const float*)d_in[1];
  const float* conv1_w  = (const float*)d_in[2];
  const float* conv2_w  = (const float*)d_in[3];
  const float* conv3_w  = (const float*)d_in[4];
  const float* conv4_w  = (const float*)d_in[5];
  const float* conv1_b  = (const float*)d_in[6];
  const float* conv2_b  = (const float*)d_in[7];
  const float* conv3_b  = (const float*)d_in[8];
  const float* conv4_b  = (const float*)d_in[9];
  const float* pos_w    = (const float*)d_in[10];
  const float* pos_b    = (const float*)d_in[11];
  const float* mlp_w1   = (const float*)d_in[12];
  const float* mlp_b1   = (const float*)d_in[13];
  const float* mlp_w2   = (const float*)d_in[14];
  const float* mlp_b2   = (const float*)d_in[15];
  const float* ln_in_g  = (const float*)d_in[16];
  const float* ln_in_b  = (const float*)d_in[17];
  const float* ln_sl_g  = (const float*)d_in[18];
  const float* ln_sl_b  = (const float*)d_in[19];
  const float* ln_ff_g  = (const float*)d_in[20];
  const float* ln_ff_b  = (const float*)d_in[21];
  const float* slots_mu = (const float*)d_in[22];
  const float* slots_ls = (const float*)d_in[23];
  const float* wq       = (const float*)d_in[24];
  const float* wk       = (const float*)d_in[25];
  const float* wv       = (const float*)d_in[26];
  const float* wi_w     = (const float*)d_in[27];
  const float* wi_b     = (const float*)d_in[28];
  const float* ws_w     = (const float*)d_in[29];
  const float* ws_b     = (const float*)d_in[30];
  const float* gru_wih  = (const float*)d_in[31];
  const float* gru_whh  = (const float*)d_in[32];
  const float* gru_bih  = (const float*)d_in[33];
  const float* gru_bhh  = (const float*)d_in[34];
  const float* fc1_w    = (const float*)d_in[35];
  const float* fc1_b    = (const float*)d_in[36];
  const float* fc2_w    = (const float*)d_in[37];
  const float* fc2_b    = (const float*)d_in[38];

  char* P = (char*)d_ws;
  size_t off = 0;
  auto ALLOC = [&](size_t nbytes) -> char* {
    char* p = P + off;
    off += (nbytes + 255) & ~(size_t)255;
    return p;
  };
  _Float16* h1h     = (_Float16*)ALLOC(16777216ull * 2);
  _Float16* h2h     = (_Float16*)ALLOC(4194304ull * 2);
  _Float16* h3h     = (_Float16*)ALLOC(1048576ull * 2);
  _Float16* h4h     = (_Float16*)ALLOC(1048576ull * 2);
  _Float16* x16     = (_Float16*)ALLOC(786432ull * 2);
  _Float16* wc1h    = (_Float16*)ALLOC(64 * 96 * 2);
  _Float16* wc2h    = (_Float16*)ALLOC(64 * 1600 * 2);
  _Float16* wc3h    = (_Float16*)ALLOC(64 * 1600 * 2);
  _Float16* wc4h    = (_Float16*)ALLOC(64 * 1600 * 2);
  _Float16* bt_mlp1 = (_Float16*)ALLOC(64 * 64 * 2);
  _Float16* bt_mlp2 = (_Float16*)ALLOC(64 * 64 * 2);
  _Float16* bt_wq   = (_Float16*)ALLOC(64 * 64 * 2);
  _Float16* bt_wk   = (_Float16*)ALLOC(64 * 64 * 2);
  _Float16* bt_wv   = (_Float16*)ALLOC(64 * 64 * 2);
  _Float16* bt_wih  = (_Float16*)ALLOC(192 * 64 * 2);
  _Float16* bt_whh  = (_Float16*)ALLOC(192 * 64 * 2);
  _Float16* bt_fc1  = (_Float16*)ALLOC(128 * 64 * 2);
  _Float16* bt_fc2  = (_Float16*)ALLOC(64 * 128 * 2);
  _Float16* feats0h = (_Float16*)ALLOC(1048576ull * 2);
  _Float16* t1h     = (_Float16*)ALLOC(1048576ull * 2);
  float*    feats   = (float*)   ALLOC(1048576ull * 4);
  _Float16* inp_h   = (_Float16*)ALLOC(1048576ull * 2);
  float*    kbuf    = (float*)   ALLOC(1048576ull * 4);
  _Float16* vT      = (_Float16*)ALLOC(1048576ull * 2);
  float*    logitsN = (float*)   ALLOC(16384 * 4);
  float*    abuf    = (float*)   ALLOC(16384 * 4);
  float*    slog    = (float*)   ALLOC(112 * 4);
  float*    bmbuf   = (float*)   ALLOC(112 * 4);
  float*    slots   = (float*)   ALLOC(7168 * 4);
  _Float16* slots_h = (_Float16*)ALLOC(7168 * 2);
  _Float16* sbuf_h  = (_Float16*)ALLOC(7168 * 2);
  float*    qbuf    = (float*)   ALLOC(7168 * 4);
  float*    attn    = (float*)   ALLOC(114688ull * 4);
  _Float16* upd_h   = (_Float16*)ALLOC(7168 * 2);
  float*    slots2  = (float*)   ALLOC(7168 * 4);
  _Float16* snbuf_h = (_Float16*)ALLOC(7168 * 2);
  _Float16* t2h     = (_Float16*)ALLOC(14336 * 2);
  float*    gi      = (float*)   ALLOC(21504 * 4);
  float*    gh      = (float*)   ALLOC(21504 * 4);

  // ---- prep: f16 weights/input ----
  cvt_f32_f16_kernel<<<3072, 256, 0, stream>>>(x, x16, 786432);
  prep_convw_kernel<<<(64 * 96 + 255) / 256, 256, 0, stream>>>(conv1_w, wc1h, 75, 96);
  prep_convw_kernel<<<(64 * 1600 + 255) / 256, 256, 0, stream>>>(conv2_w, wc2h, 1600, 1600);
  prep_convw_kernel<<<(64 * 1600 + 255) / 256, 256, 0, stream>>>(conv3_w, wc3h, 1600, 1600);
  prep_convw_kernel<<<(64 * 1600 + 255) / 256, 256, 0, stream>>>(conv4_w, wc4h, 1600, 1600);
  prep_bt_kernel<<<16, 256, 0, stream>>>(mlp_w1, bt_mlp1, 64, 64, 0);
  prep_bt_kernel<<<16, 256, 0, stream>>>(mlp_w2, bt_mlp2, 64, 64, 0);
  prep_bt_kernel<<<16, 256, 0, stream>>>(wq, bt_wq, 64, 64, 0);
  prep_bt_kernel<<<16, 256, 0, stream>>>(wk, bt_wk, 64, 64, 0);
  prep_bt_kernel<<<16, 256, 0, stream>>>(wv, bt_wv, 64, 64, 0);
  prep_bt_kernel<<<48, 256, 0, stream>>>(gru_wih, bt_wih, 192, 64, 1);
  prep_bt_kernel<<<48, 256, 0, stream>>>(gru_whh, bt_whh, 192, 64, 1);
  prep_bt_kernel<<<32, 256, 0, stream>>>(fc1_w, bt_fc1, 128, 64, 0);
  prep_bt_kernel<<<32, 256, 0, stream>>>(fc2_w, bt_fc2, 64, 128, 0);

  // ---- encoder CNN (f16) ----
  conv_relu_wmma<3, 1><<<4096, 256, 0, stream>>>(x16, wc1h, conv1_b, h1h, NB, 128, 128, 128, 128);
  conv_relu_wmma<64, 2><<<1024, 256, 0, stream>>>(h1h, wc2h, conv2_b, h2h, NB, 128, 128, 64, 64);
  conv_relu_wmma<64, 2><<<256, 256, 0, stream>>>(h2h, wc3h, conv3_b, h3h, NB, 64, 64, 32, 32);
  conv_relu_wmma<64, 1><<<256, 256, 0, stream>>>(h3h, wc4h, conv4_b, h4h, NB, 32, 32, 32, 32);

  pos_embed_kernel<<<(NB * NPIX * DIM + 255) / 256, 256, 0, stream>>>(h4h, pos_w, pos_b, feats0h);
  gemm_wmma<64, 64, 1, 1><<<128, 256, 0, stream>>>(feats0h, bt_mlp1, mlp_b1, nullptr, nullptr, t1h, 16384);
  gemm_wmma<64, 64, 0, 0><<<128, 256, 0, stream>>>(t1h, bt_mlp2, mlp_b2, nullptr, feats, nullptr, 16384);

  ln_rows_kernel<<<2048, 256, 0, stream>>>(feats, ln_in_g, ln_in_b, wi_w, wi_b, inp_h, logitsN, 16384);
  softmax_scale_kernel<<<NB, 256, 0, stream>>>(logitsN, abuf, NPIX, (float)NSLOT);
  gemm_wmma<64, 64, 0, 0><<<128, 256, 0, stream>>>(inp_h, bt_wk, nullptr, nullptr, kbuf, nullptr, 16384);
  gemm_wmma<64, 64, 3, 0><<<128, 256, 0, stream>>>(inp_h, bt_wv, nullptr, nullptr, nullptr, vT, 16384);
  slots_init_kernel<<<28, 256, 0, stream>>>(slots_mu, slots_ls, noise, slots, slots_h);

  for (int iter = 0; iter < 3; ++iter) {
    ln_rows_kernel<<<14, 256, 0, stream>>>(slots, ln_sl_g, ln_sl_b, ws_w, ws_b, sbuf_h, slog, NB * NSLOT);
    softmax_scale_kernel<<<NB, 256, 0, stream>>>(slog, bmbuf, NSLOT, (float)NSLOT);
    gemm_wmma<64, 64, 0, 0><<<1, 256, 0, stream>>>(sbuf_h, bt_wq, nullptr, nullptr, qbuf, nullptr, 112);
    mesh_sinkhorn_kernel<<<NB, 256, 0, stream>>>(kbuf, qbuf, abuf, bmbuf, attn);
    attn_v_wmma<<<NB, 256, 0, stream>>>(attn, vT, upd_h);
    gemm_wmma<64, 192, 0, 0><<<1, 256, 0, stream>>>(upd_h, bt_wih, gru_bih, nullptr, gi, nullptr, 112);
    gemm_wmma<64, 192, 0, 0><<<1, 256, 0, stream>>>(slots_h, bt_whh, gru_bhh, nullptr, gh, nullptr, 112);
    gru_kernel<<<28, 256, 0, stream>>>(gi, gh, slots, slots2);
    ln_rows_kernel<<<14, 256, 0, stream>>>(slots2, ln_ff_g, ln_ff_b, nullptr, nullptr, snbuf_h, nullptr, NB * NSLOT);
    gemm_wmma<64, 128, 1, 1><<<1, 256, 0, stream>>>(snbuf_h, bt_fc1, fc1_b, nullptr, nullptr, t2h, 112);
    gemm_wmma<128, 64, 2, 0><<<1, 256, 0, stream>>>(t2h, bt_fc2, fc2_b, slots2, slots, slots_h, 112);
  }

  pack_kernel<<<(NB * NSLOT * (DIM + NPIX) + 255) / 256, 256, 0, stream>>>(slots, attn, (float*)d_out);
}
